// ChannelMix_26774826123920
// MI455X (gfx1250) — 8-bit matrix operand order repaired and checked by reading, compiled, not run on hardware
//
#include <hip/hip_runtime.h>
#include <hip/hip_bf16.h>

typedef __attribute__((ext_vector_type(8))) int v8i;

#ifndef LL
#define LL long long
#endif

__device__ __forceinline__ float block_reduce_sum_256(float v, float* red) {
    int t = threadIdx.x;
    red[t] = v;
    __syncthreads();
    #pragma unroll
    for (int off = 128; off > 0; off >>= 1) {
        if (t < off) red[t] += red[t + off];
        __syncthreads();
    }
    float r = red[0];
    __syncthreads();
    return r;
}

__global__ __launch_bounds__(256)
void abs_sum_partial(const float* __restrict__ w, LL n, float* __restrict__ partial) {
    __shared__ float red[256];
    float s = 0.f;
    for (LL i = (LL)blockIdx.x * blockDim.x + threadIdx.x; i < n;
         i += (LL)gridDim.x * blockDim.x)
        s += fabsf(w[i]);
    s = block_reduce_sum_256(s, red);
    if (threadIdx.x == 0) partial[blockIdx.x] = s;
}

__global__ __launch_bounds__(256)
void finalize_wscale(const float* __restrict__ partial, int nparts, LL n,
                     float* __restrict__ scale_out) {
    __shared__ float red[256];
    float s = 0.f;
    for (int i = threadIdx.x; i < nparts; i += 256) s += partial[i];
    s = block_reduce_sum_256(s, red);
    if (threadIdx.x == 0) *scale_out = fmaxf(s / (float)n, 1e-8f);
}

__global__ __launch_bounds__(256)
void weight_quant(const float* __restrict__ w, const float* __restrict__ scale_p,
                  signed char* __restrict__ q, LL n) {
    float inv = 1.f / *scale_p;
    for (LL i = (LL)blockIdx.x * blockDim.x + threadIdx.x; i < n;
         i += (LL)gridDim.x * blockDim.x) {
        float v = fminf(fmaxf(w[i] * inv, -1.f), 1.f);
        q[i] = (signed char)(int)rintf(v);
    }
}

template <bool SHIFT>
__global__ __launch_bounds__(256)
void act_prep(const float* __restrict__ in, const float* __restrict__ mu,
              const float* __restrict__ g, const float* __restrict__ bb,
              int Dlen, int T,
              signed char* __restrict__ q, float* __restrict__ scales) {
    __shared__ float yrow[8192];
    __shared__ float red[256];

    const int m = blockIdx.x;
    const float* row = in + (LL)m * Dlen;
    const float* prow = row - Dlen;
    bool has_prev = false;
    if (SHIFT) has_prev = ((m % T) != 0);

    float s1 = 0.f;
    for (int d = threadIdx.x; d < Dlen; d += 256) {
        float xv = row[d];
        float y;
        if (SHIFT) {
            float xp = has_prev ? prow[d] : 0.f;
            y = xv + (xp - xv) * mu[d];
        } else {
            y = xv;
        }
        yrow[d] = y;
        s1 += y;
    }
    float mean = block_reduce_sum_256(s1, red) / (float)Dlen;

    float s2 = 0.f;
    for (int d = threadIdx.x; d < Dlen; d += 256) {
        float dl = yrow[d] - mean;
        s2 += dl * dl;
    }
    float var = block_reduce_sum_256(s2, red) / (float)Dlen;
    float rstd = rsqrtf(var + 1e-5f);

    float sa = 0.f;
    for (int d = threadIdx.x; d < Dlen; d += 256) {
        float ln = (yrow[d] - mean) * rstd * g[d] + bb[d];
        yrow[d] = ln;
        sa += fabsf(ln);
    }
    float meanabs = block_reduce_sum_256(sa, red) / (float)Dlen;
    float scale = fmaxf(meanabs, 1e-8f) * 2.5f / 127.f;
    float inv = 1.f / scale;

    signed char* qrow = q + (LL)m * Dlen;
    for (int d = threadIdx.x; d < Dlen; d += 256) {
        float xs = fminf(fmaxf(yrow[d] * inv, -127.f), 127.f);
        qrow[d] = (signed char)(int)rintf(xs);
    }
    if (threadIdx.x == 0) scales[m] = scale;
}

template <int ROWS>
__device__ __forceinline__ void stage_async(const signed char* __restrict__ g,
                                            int ldk, int k0,
                                            const signed char* lds_base) {
    const unsigned base = (unsigned)(size_t)lds_base;
    #pragma unroll
    for (int i = 0; i < (ROWS * 16) / 256; ++i) {
        int idx = threadIdx.x + i * 256;
        int r = idx >> 4, dw = idx & 15;
        unsigned long long ga =
            (unsigned long long)(size_t)(g + (LL)r * ldk + k0 + dw * 4);
        unsigned la = base + r * 68 + dw * 4;
        asm volatile("global_load_async_to_lds_b32 %0, %1, off"
                     :: "v"(la), "v"(ga)
                     : "memory");
    }
}

template <int MODE>
__global__ __launch_bounds__(256)
void gemm_iu8(const signed char* __restrict__ Aq, const signed char* __restrict__ Bq,
              const float* __restrict__ sa, const float* __restrict__ swp,
              int Nsz, int Ksz,
              const float* __restrict__ rg, float* __restrict__ Cout) {
    __shared__ signed char At[2][64 * 68];
    __shared__ signed char Bt[2][128 * 68];

    const int tid = threadIdx.x;
    const int lane = tid & 31;
    const int wv = tid >> 5;
    const int wm = (wv >> 2) * 32;
    const int wn = (wv & 3) * 32;
    const int m0 = blockIdx.y * 64;
    const int n0 = blockIdx.x * 128;
    const int half = lane >> 4;
    const int l15 = lane & 15;

    const signed char* Abase = Aq + (LL)m0 * Ksz;
    const signed char* Bbase = Bq + (LL)n0 * Ksz;

    v8i acc[2][2] = {};

    const int nK = Ksz >> 6;

    stage_async<64>(Abase, Ksz, 0, At[0]);
    stage_async<128>(Bbase, Ksz, 0, Bt[0]);

    for (int t = 0; t < nK; ++t) {
        const int cur = t & 1;

        if (t + 1 < nK) {
            const int k1 = (t + 1) << 6;
            stage_async<64>(Abase, Ksz, k1, At[cur ^ 1]);
            stage_async<128>(Bbase, Ksz, k1, Bt[cur ^ 1]);
            if (t + 2 < nK) {
                const int k2 = (t + 2) << 6;
                __builtin_prefetch(Abase + (LL)(tid >> 2) * Ksz + k2 + (tid & 3) * 16, 0, 0);
            }
            asm volatile("s_wait_asynccnt 12" ::: "memory");
        } else {
            asm volatile("s_wait_asynccnt 0" ::: "memory");
        }
        __syncthreads();

        v8i af[2], bf[2];
        #pragma unroll
        for (int f = 0; f < 2; ++f) {
            const int arow = wm + f * 16 + l15;
            const int brow = wn + f * 16 + l15;
            #pragma unroll
            for (int v = 0; v < 8; ++v) {
                int ka = ((v >> 2) << 5) + (((v >> 1) & 1) << 4) + (half << 3) + ((v & 1) << 2);
                af[f][v] = *reinterpret_cast<const int*>(&At[cur][arow * 68 + ka]);
                int kb = ((v >> 2) << 5) + (((v >> 1) & 1) << 4) + (half << 3) + ((v & 1) << 2);
                bf[f][v] = *reinterpret_cast<const int*>(&Bt[cur][brow * 68 + kb]);
            }
        }

        #pragma unroll
        for (int fm = 0; fm < 2; ++fm)
            #pragma unroll
            for (int fn = 0; fn < 2; ++fn)
                acc[fm][fn] = __builtin_amdgcn_wmma_i32_16x16x64_iu8(
                     true, af[fm],  true, bf[fn],
                    acc[fm][fn],  false,  false);

        __syncthreads();
    }

    const float sw = *swp;
    #pragma unroll
    for (int fm = 0; fm < 2; ++fm) {
        #pragma unroll
        for (int fn = 0; fn < 2; ++fn) {
            #pragma unroll
            for (int v = 0; v < 8; ++v) {
                int mrow = m0 + wm + fm * 16 + v + half * 8;
                int ncol = n0 + wn + fn * 16 + l15;
                float val = (float)acc[fm][fn][v] * sa[mrow] * sw;
                LL idx = (LL)mrow * Nsz + ncol;
                if (MODE == 0) {
                    float p = fmaxf(val, 0.f);
                    Cout[idx] = p * p;
                } else if (MODE == 1) {
                    Cout[idx] = val;
                } else {
                    float gt = 1.f / (1.f + expf(-rg[idx]));
                    Cout[idx] = gt * val;
                }
            }
        }
    }
}

__global__ __launch_bounds__(256)
void copy_state(const float* __restrict__ x, float* __restrict__ dst,
                int Dlen, int T, int Bn) {
    int i = blockIdx.x * blockDim.x + threadIdx.x;
    if (i < Bn * Dlen) {
        int b = i / Dlen, d = i % Dlen;
        dst[i] = x[(LL)(b * T + T - 1) * Dlen + d];
    }
}

extern "C" void kernel_launch(void* const* d_in, const int* in_sizes, int n_in,
                              void* d_out, int out_size, void* d_ws, size_t ws_size,
                              hipStream_t stream) {
    const float* x    = (const float*)d_in[0];
    const float* mu_k = (const float*)d_in[1];
    const float* mu_r = (const float*)d_in[2];
    const float* wk   = (const float*)d_in[3];
    const float* gk   = (const float*)d_in[4];
    const float* bk   = (const float*)d_in[5];
    const float* wr   = (const float*)d_in[6];
    const float* gr   = (const float*)d_in[7];
    const float* br   = (const float*)d_in[8];
    const float* wv   = (const float*)d_in[9];
    const float* gv   = (const float*)d_in[10];
    const float* bv   = (const float*)d_in[11];
    float* out = (float*)d_out;

    const int D = in_sizes[4];
    const int H = in_sizes[3] / D;
    const int M = in_sizes[0] / D;
    const int Bn = out_size / D - M;
    const int T = M / Bn;

    size_t off = 0;
    auto alloc = [&](size_t bytes) -> char* {
        char* p = (char*)d_ws + off;
        off = (off + bytes + 255) & ~(size_t)255;
        return p;
    };
    signed char* wkq = (signed char*)alloc((size_t)H * D);
    signed char* wrq = (signed char*)alloc((size_t)D * D);
    signed char* wvq = (signed char*)alloc((size_t)D * H);
    signed char* xkq = (signed char*)alloc((size_t)M * D);
    signed char* xrq = (signed char*)alloc((size_t)M * D);
    signed char* kq  = (signed char*)alloc((size_t)M * H);
    float* sak = (float*)alloc((size_t)M * 4);
    float* sar = (float*)alloc((size_t)M * 4);
    float* sav = (float*)alloc((size_t)M * 4);
    float* swk = (float*)alloc(256);
    float* swr = (float*)alloc(256);
    float* swv = (float*)alloc(256);
    float* partial = (float*)alloc(1024 * 4);
    float* kact = (float*)alloc((size_t)M * H * 4);
    float* rgbf = (float*)alloc((size_t)M * D * 4);
    (void)ws_size;

    abs_sum_partial<<<1024, 256, 0, stream>>>(wk, (LL)H * D, partial);
    finalize_wscale<<<1, 256, 0, stream>>>(partial, 1024, (LL)H * D, swk);
    weight_quant<<<2048, 256, 0, stream>>>(wk, swk, wkq, (LL)H * D);

    abs_sum_partial<<<1024, 256, 0, stream>>>(wr, (LL)D * D, partial);
    finalize_wscale<<<1, 256, 0, stream>>>(partial, 1024, (LL)D * D, swr);
    weight_quant<<<2048, 256, 0, stream>>>(wr, swr, wrq, (LL)D * D);

    abs_sum_partial<<<1024, 256, 0, stream>>>(wv, (LL)D * H, partial);
    finalize_wscale<<<1, 256, 0, stream>>>(partial, 1024, (LL)D * H, swv);
    weight_quant<<<2048, 256, 0, stream>>>(wv, swv, wvq, (LL)D * H);

    act_prep<true><<<M, 256, 0, stream>>>(x, mu_k, gk, bk, D, T, xkq, sak);
    act_prep<true><<<M, 256, 0, stream>>>(x, mu_r, gr, br, D, T, xrq, sar);

    {
        dim3 grid(H / 128, M / 64);
        gemm_iu8<0><<<grid, 256, 0, stream>>>(xkq, wkq, sak, swk, H, D, nullptr, kact);
    }

    act_prep<false><<<M, 256, 0, stream>>>(kact, nullptr, gv, bv, H, T, kq, sav);

    {
        dim3 grid(D / 128, M / 64);
        gemm_iu8<1><<<grid, 256, 0, stream>>>(xrq, wrq, sar, swr, D, D, nullptr, rgbf);
    }

    {
        dim3 grid(D / 128, M / 64);
        gemm_iu8<2><<<grid, 256, 0, stream>>>(kq, wvq, sav, swv, D, H, rgbf, out);
    }

    copy_state<<<(Bn * D + 255) / 256, 256, 0, stream>>>(
        x, out + (LL)M * D, D, T, Bn);
}
